// EfficientAttention_11785390260492
// MI455X (gfx1250) — matrix operand order repaired and checked by reading, not compiled, not run on hardware
//
#include <hip/hip_runtime.h>
#include <hip/hip_bf16.h>

typedef __bf16 bf16_t;
typedef __attribute__((ext_vector_type(16))) __bf16 v16bf;
typedef __attribute__((ext_vector_type(8)))  __bf16 v8bf;
typedef __attribute__((ext_vector_type(8)))  float  v8f;

#define B_SZ   2
#define T_SZ   2048
#define DM     2048
#define NH     16
#define NKV    4
#define DH     128
#define WIN    512
#define NROWS  (B_SZ * T_SZ)       // 4096
#define NQKV   (DM + 2 * NKV * DH) // 3072
#define KSTEP  64                  // GEMM K-slice staged in LDS

// ---------------------------------------------------------------------------
// WMMA fragment helpers (CDNA5 16x16x32 bf16 layouts, ISA 7.12.2)
// A 16x32: lane L -> row m=L&15, half hf=L>>4 covers k = hf*8..hf*8+7 (vgpr0-3)
//          and 16+hf*8..16+hf*8+7 (vgpr4-7)  => two contiguous 8-elt runs.
// B 32x16: lane L -> col n=L&15, kg=L>>4 covers k = kg*8..kg*8+7 and 16+kg*8..16+kg*8+7.
// ---------------------------------------------------------------------------
__device__ __forceinline__ v16bf load_a_frag(const bf16_t* row, int hf) {
  union { v16bf v; v8bf p[2]; } u;
  u.p[0] = *(const v8bf*)(row + hf * 8);
  u.p[1] = *(const v8bf*)(row + 16 + hf * 8);
  return u.v;
}

__device__ __forceinline__ v16bf load_b_frag(const bf16_t* row, int kg) {
  union { v16bf v; v8bf p[2]; } u;
  const bf16_t* p = row + kg * 8;
  u.p[0] = *(const v8bf*)(p);
  u.p[1] = *(const v8bf*)(p + 16);
  return u.v;
}

__device__ __forceinline__ v8f wmma_bf16(v16bf a, v16bf b, v8f c) {
  return __builtin_amdgcn_wmma_f32_16x16x32_bf16(false, a, false, b,
                                                 (short)0, c, false, false);
}

// gfx1250 async copy global->LDS (ASYNCcnt-tracked, ISA 08 section 4):
// per lane: LDS[lds_addr] = MEM[gaddr], 16 bytes.
__device__ __forceinline__ void async_copy_b128(unsigned lds_addr,
                                                const bf16_t* gptr) {
  asm volatile("global_load_async_to_lds_b128 %0, %1, off"
               :: "v"(lds_addr), "v"((unsigned long long)(uintptr_t)gptr)
               : "memory");
}

// ---------------------------------------------------------------------------
// Conversion / packing kernels
// ---------------------------------------------------------------------------
__global__ void convert_x_kernel(const float* __restrict__ x,
                                 bf16_t* __restrict__ xb, int n) {
  int i = blockIdx.x * blockDim.x + threadIdx.x;
  if (i < n) xb[i] = (bf16_t)x[i];
}

// Wqkv_t[n][c] = W*[c][n], n in [0,3072) fused q|k|v, c in [0,2048)
__global__ void pack_wqkv_kernel(const float* __restrict__ Wq,
                                 const float* __restrict__ Wk,
                                 const float* __restrict__ Wv,
                                 bf16_t* __restrict__ Wt) {
  int idx = blockIdx.x * blockDim.x + threadIdx.x;
  if (idx >= NQKV * DM) return;
  int n = idx / DM, c = idx % DM;
  float v;
  if (n < DM)            v = Wq[(size_t)c * DM  + n];
  else if (n < DM + 512) v = Wk[(size_t)c * 512 + (n - DM)];
  else                   v = Wv[(size_t)c * 512 + (n - DM - 512)];
  Wt[idx] = (bf16_t)v;
}

__global__ void transpose_wo_kernel(const float* __restrict__ Wo,
                                    bf16_t* __restrict__ Wt) {
  int idx = blockIdx.x * blockDim.x + threadIdx.x;
  if (idx >= DM * DM) return;
  int n = idx / DM, c = idx % DM;
  Wt[idx] = (bf16_t)Wo[(size_t)c * DM + n];
}

// ---------------------------------------------------------------------------
// Block-cooperative bf16 WMMA GEMM with async double-buffered LDS staging.
//   C[M][N] (f32) = A[M][Kd] (bf16 row major) x Bt[N][Kd] (bf16, == B^T)
// Workgroup: 256 threads = 8 waves, tile 128x128, K-slice 64.
// Wave (wm,wn) in 4x2 grid owns a 32x64 subtile -> 2x4 = 8 accumulators,
// 16 wmma per slice between barriers.
// LDS: 2 buffers x (A 128x64 + Bt 128x64) bf16 = 64 KB.
// ---------------------------------------------------------------------------
__global__ void __launch_bounds__(256)
gemm_bf16_kernel(const bf16_t* __restrict__ A, const bf16_t* __restrict__ Bt,
                 float* __restrict__ C, int M, int N, int Kd) {
  __shared__ __align__(16) bf16_t lds_a[2][128][KSTEP];
  __shared__ __align__(16) bf16_t lds_b[2][128][KSTEP];

  const int tid  = threadIdx.x;
  const int lane = tid & 31;
  const int w    = tid >> 5;
  const int hf = lane >> 4, nl = lane & 15;

  const int tiles_n = N >> 7;
  const int tm = blockIdx.x / tiles_n;
  const int tn = blockIdx.x % tiles_n;

  const int wm = w & 3;   // 4 M-subwaves * 32 rows
  const int wn = w >> 2;  // 2 N-subwaves * 64 cols

  const bf16_t* agmem = A  + (size_t)(tm * 128) * Kd;
  const bf16_t* bgmem = Bt + (size_t)(tn * 128) * Kd;

  // copy plan: 1024 chunks of 16B per 128x64 tile; thread owns 4 chunks,
  // rows r0 + {0,32,64,96} at element-column c0 (LDS row stride 128 bytes).
  const int r0 = tid >> 3;
  const int c0 = (tid & 7) * 8;
  const unsigned la = (unsigned)(uintptr_t)&lds_a[0][0][0];
  const unsigned lb = (unsigned)(uintptr_t)&lds_b[0][0][0];
  const unsigned bufb = 128 * KSTEP * 2; // 16 KB per buffer

  auto issue = [&](int buf, int ks) {
#pragma unroll
    for (int i = 0; i < 4; ++i) {
      int row = r0 + i * 32;
      const bf16_t* ga = agmem + (size_t)row * Kd + ks + c0;
      const bf16_t* gb = bgmem + (size_t)row * Kd + ks + c0;
      unsigned da = la + buf * bufb + row * (KSTEP * 2) + c0 * 2;
      unsigned db = lb + buf * bufb + row * (KSTEP * 2) + c0 * 2;
      __builtin_prefetch(ga + KSTEP, 0, 1); // warm L2 one slice further ahead
      __builtin_prefetch(gb + KSTEP, 0, 1);
      async_copy_b128(da, ga);
      async_copy_b128(db, gb);
    }
  }; // 8 async instructions per wave per slice

  v8f acc[2][4] = {};

  auto compute = [&](int buf) {
#pragma unroll
    for (int kc = 0; kc < KSTEP; kc += 32) {
      v16bf af[2], bfr[4];
#pragma unroll
      for (int mi = 0; mi < 2; ++mi)
        af[mi] = load_a_frag(&lds_a[buf][wm * 32 + mi * 16 + nl][kc], hf);
#pragma unroll
      for (int ni = 0; ni < 4; ++ni)
        bfr[ni] = load_b_frag(&lds_b[buf][wn * 64 + ni * 16 + nl][kc], hf);
#pragma unroll
      for (int mi = 0; mi < 2; ++mi)
#pragma unroll
        for (int ni = 0; ni < 4; ++ni)
          acc[mi][ni] = wmma_bf16(af[mi], bfr[ni], acc[mi][ni]);
    }
  };

  const int nslices = Kd / KSTEP;
  issue(0, 0);
  // steady state: no branches, issue next slice then overlap with 16 wmma
  for (int s = 0; s < nslices - 1; ++s) {
    int buf = s & 1;
    issue(buf ^ 1, (s + 1) * KSTEP);
    asm volatile("s_wait_asynccnt 8" ::: "memory"); // slice s landed
    __syncthreads();
    compute(buf);
    __syncthreads();
  }
  // epilogue: drain the last slice
  asm volatile("s_wait_asynccnt 0" ::: "memory");
  __syncthreads();
  compute((nslices - 1) & 1);

#pragma unroll
  for (int mi = 0; mi < 2; ++mi)
#pragma unroll
    for (int ni = 0; ni < 4; ++ni)
#pragma unroll
      for (int r = 0; r < 8; ++r) {
        size_t row = (size_t)(tm * 128 + wm * 32 + mi * 16 + r + 8 * hf) * N;
        int col = tn * 128 + wn * 64 + ni * 16 + nl;
        C[row + col] = acc[mi][ni][r];
      }
}

// ---------------------------------------------------------------------------
// RoPE epilogues: QKV f32 [4096][3072] -> bf16 tensors in attention layouts
// ---------------------------------------------------------------------------
__global__ void rope_q_kernel(const float* __restrict__ QKV,
                              const float* __restrict__ cosb,
                              const float* __restrict__ sinb,
                              bf16_t* __restrict__ qb) {
  int idx = blockIdx.x * blockDim.x + threadIdx.x; // B*T*NH*64
  if (idx >= B_SZ * T_SZ * NH * 64) return;
  int d2 = idx & 63;
  int h  = (idx >> 6) & 15;
  int t  = (idx >> 10) & (T_SZ - 1);
  int b  = idx >> 21;
  size_t row = (size_t)(b * T_SZ + t) * NQKV + h * DH;
  float qr = QKV[row + 2 * d2], qi = QKV[row + 2 * d2 + 1];
  float c = cosb[t * 64 + d2], s = sinb[t * 64 + d2];
  size_t o = ((size_t)(b * NH + h) * T_SZ + t) * DH + 2 * d2;
  qb[o]     = (bf16_t)(qr * c - qi * s);
  qb[o + 1] = (bf16_t)(qr * s + qi * c);
}

__global__ void rope_k_kernel(const float* __restrict__ QKV,
                              const float* __restrict__ cosb,
                              const float* __restrict__ sinb,
                              bf16_t* __restrict__ kb) {
  int idx = blockIdx.x * blockDim.x + threadIdx.x; // B*T*NKV*64
  if (idx >= B_SZ * T_SZ * NKV * 64) return;
  int d2 = idx & 63;
  int h  = (idx >> 6) & 3;
  int t  = (idx >> 8) & (T_SZ - 1);
  int b  = idx >> 19;
  size_t row = (size_t)(b * T_SZ + t) * NQKV + DM + h * DH;
  float kr = QKV[row + 2 * d2], ki = QKV[row + 2 * d2 + 1];
  float c = cosb[t * 64 + d2], s = sinb[t * 64 + d2];
  size_t o = ((size_t)(b * NKV + h) * T_SZ + t) * DH + 2 * d2;
  kb[o]     = (bf16_t)(kr * c - ki * s);
  kb[o + 1] = (bf16_t)(kr * s + ki * c);
}

// Vt[b][kv][d][t] (transposed) so P*V B-fragments are contiguous loads.
__global__ void conv_v_kernel(const float* __restrict__ QKV,
                              bf16_t* __restrict__ vt) {
  int idx = blockIdx.x * blockDim.x + threadIdx.x; // B*T*NKV*128
  if (idx >= B_SZ * T_SZ * NKV * DH) return;
  int d = idx & 127;
  int h = (idx >> 7) & 3;
  int t = (idx >> 9) & (T_SZ - 1);
  int b = idx >> 20;
  float v = QKV[(size_t)(b * T_SZ + t) * NQKV + DM + 512 + h * DH + d];
  vt[((size_t)(b * NKV + h) * DH + d) * T_SZ + t] = (bf16_t)v;
}

// ---------------------------------------------------------------------------
// Flash-style sliding-window attention.
// One wave per (b, head, 16-query tile); 32 keys per iteration.
// S: 8x v_wmma (2 key subtiles x 4 d-chunks); P*V: 8x v_wmma (8 d-col tiles).
// ---------------------------------------------------------------------------
__global__ void __launch_bounds__(256)
attn_kernel(const bf16_t* __restrict__ qb, const bf16_t* __restrict__ kb,
            const bf16_t* __restrict__ vt, bf16_t* __restrict__ yb) {
  __shared__ __align__(16) bf16_t plds[8][16][32]; // per-wave P staging

  int lane = threadIdx.x & 31;
  int w    = threadIdx.x >> 5;
  int wid  = blockIdx.x * 8 + w;            // 4096 waves total
  int qt   = wid & 127;
  int head = (wid >> 7) & (NH - 1);
  int b    = wid >> 11;
  int q0   = qt * 16;
  int kvh  = head >> 2;                     // GQA: 4 Q heads per KV head
  int hf = lane >> 4, nl = lane & 15;

  const bf16_t* qrow = qb + (size_t)(b * NH + head) * T_SZ * DH;
  const bf16_t* krow = kb + (size_t)(b * NKV + kvh) * T_SZ * DH;
  const bf16_t* vrow = vt + (size_t)(b * NKV + kvh) * DH * T_SZ;
  const bf16_t* qlane = qrow + (size_t)(q0 + nl) * DH;

  v8f o[8] = {};
  float mrow[8], lrow[8];
#pragma unroll
  for (int r = 0; r < 8; ++r) { mrow[r] = -3.0e38f; lrow[r] = 0.0f; }

  const float scale = 0.08838834764831845f; // 1/sqrt(128)
  int kstart = q0 - (WIN - 1);
  if (kstart < 0) kstart = 0;
  kstart &= ~31;
  int kend = q0 + 15;

  for (int k0 = kstart; k0 <= kend; k0 += 32) {
    // ---- S = Q K^T for 32 keys -----------------------------------------
    v8f s0 = {}, s1 = {};
    int j0 = k0 + nl;       if (j0 > T_SZ - 1) j0 = T_SZ - 1;
    int j1 = k0 + 16 + nl;  if (j1 > T_SZ - 1) j1 = T_SZ - 1;
    const bf16_t* k0row = krow + (size_t)j0 * DH;
    const bf16_t* k1row = krow + (size_t)j1 * DH;
#pragma unroll
    for (int kc = 0; kc < DH; kc += 32) {
      v16bf aq  = load_a_frag(qlane + kc, hf);
      v16bf bk0 = load_b_frag(k0row + kc, hf);
      v16bf bk1 = load_b_frag(k1row + kc, hf);
      s0 = wmma_bf16(aq, bk0, s0);
      s1 = wmma_bf16(aq, bk1, s1);
    }

    // ---- mask + online softmax -----------------------------------------
    float e0a[8], e1a[8], al[8];
#pragma unroll
    for (int r = 0; r < 8; ++r) {
      int i  = q0 + r + 8 * hf;
      int c0 = k0 + nl, c1 = k0 + 16 + nl;
      bool ok0 = (c0 <= i) && (c0 > i - WIN);
      bool ok1 = (c1 <= i) && (c1 > i - WIN);
      e0a[r] = ok0 ? s0[r] * scale : -3.0e38f;
      e1a[r] = ok1 ? s1[r] * scale : -3.0e38f;
    }
#pragma unroll
    for (int r = 0; r < 8; ++r) {
      float mx = fmaxf(e0a[r], e1a[r]);
#pragma unroll
      for (int msk = 1; msk <= 8; msk <<= 1)
        mx = fmaxf(mx, __shfl_xor(mx, msk, 32));
      float mn = fmaxf(mrow[r], mx);
      float a  = __expf(mrow[r] - mn);
      float p0 = (e0a[r] < -1.0e38f) ? 0.0f : __expf(e0a[r] - mn);
      float p1 = (e1a[r] < -1.0e38f) ? 0.0f : __expf(e1a[r] - mn);
      float rs = p0 + p1;
#pragma unroll
      for (int msk = 1; msk <= 8; msk <<= 1)
        rs += __shfl_xor(rs, msk, 32);
      lrow[r] = lrow[r] * a + rs;
      mrow[r] = mn;
      al[r]   = a;
      // stage P (C-layout -> LDS row-major) for A-fragment reuse
      plds[w][r + 8 * hf][nl]      = (bf16_t)p0;
      plds[w][r + 8 * hf][16 + nl] = (bf16_t)p1;
    }
    // cross-lane handoff through LDS within the wave
    asm volatile("s_wait_dscnt 0" ::: "memory");

    // rescale accumulator rows
#pragma unroll
    for (int c = 0; c < 8; ++c)
#pragma unroll
      for (int r = 0; r < 8; ++r) o[c][r] = o[c][r] * al[r];

    // ---- O += P * V ------------------------------------------------------
    v16bf ap = load_a_frag(&plds[w][nl][0], hf);
    int kv = k0; if (kv > T_SZ - 32) kv = T_SZ - 32; // clamp (masked P == 0)
#pragma unroll
    for (int c = 0; c < 8; ++c) {
      const bf16_t* vptr = vrow + (size_t)(c * 16 + nl) * T_SZ + kv;
      v16bf bv = load_b_frag(vptr, hf);
      o[c] = wmma_bf16(ap, bv, o[c]);
    }
  }

  // ---- normalize + store y[b][t][head*128 + d] as bf16 -------------------
#pragma unroll
  for (int c = 0; c < 8; ++c)
#pragma unroll
    for (int r = 0; r < 8; ++r) {
      size_t row = (size_t)(b * T_SZ + q0 + r + 8 * hf) * DM;
      int col = head * DH + c * 16 + nl;
      yb[row + col] = (bf16_t)(o[c][r] / lrow[r]);
    }
}

// ---------------------------------------------------------------------------
// Host side
// ---------------------------------------------------------------------------
extern "C" void kernel_launch(void* const* d_in, const int* in_sizes, int n_in,
                              void* d_out, int out_size, void* d_ws, size_t ws_size,
                              hipStream_t stream) {
  const float* x    = (const float*)d_in[0];
  const float* cosb = (const float*)d_in[1];
  const float* sinb = (const float*)d_in[2];
  const float* Wq   = (const float*)d_in[3];
  const float* Wk   = (const float*)d_in[4];
  const float* Wv   = (const float*)d_in[5];
  const float* Wo   = (const float*)d_in[6];
  float* out = (float*)d_out;

  char* ws = (char*)d_ws;
  size_t off = 0;
  auto carve = [&](size_t bytes) -> void* {
    void* p = ws + off;
    off = (off + bytes + 255) & ~(size_t)255;
    return p;
  };
  bf16_t* xb    = (bf16_t*)carve((size_t)NROWS * DM * 2);
  bf16_t* wqkvt = (bf16_t*)carve((size_t)NQKV * DM * 2);
  bf16_t* wot   = (bf16_t*)carve((size_t)DM * DM * 2);
  float*  qkvf  = (float*) carve((size_t)NROWS * NQKV * 4);
  bf16_t* qbuf  = (bf16_t*)carve((size_t)B_SZ * NH * T_SZ * DH * 2);
  bf16_t* kbuf  = (bf16_t*)carve((size_t)B_SZ * NKV * T_SZ * DH * 2);
  bf16_t* vtb   = (bf16_t*)carve((size_t)B_SZ * NKV * DH * T_SZ * 2);
  bf16_t* ybuf  = (bf16_t*)carve((size_t)NROWS * DM * 2);

  const int TB = 256;
  // 1. precision conversion / weight packing
  convert_x_kernel<<<(NROWS * DM) / TB, TB, 0, stream>>>(x, xb, NROWS * DM);
  pack_wqkv_kernel<<<(NQKV * DM) / TB, TB, 0, stream>>>(Wq, Wk, Wv, wqkvt);
  transpose_wo_kernel<<<(DM * DM) / TB, TB, 0, stream>>>(Wo, wot);

  // 2. fused QKV projection: [4096,2048] x [2048,3072] -> f32 (128x128 tiles)
  gemm_bf16_kernel<<<(NROWS / 128) * (NQKV / 128), TB, 0, stream>>>(
      xb, wqkvt, qkvf, NROWS, NQKV, DM);

  // 3. RoPE + layout conversion to bf16 attention tensors
  rope_q_kernel<<<(B_SZ * T_SZ * NH * 64) / TB, TB, 0, stream>>>(qkvf, cosb, sinb, qbuf);
  rope_k_kernel<<<(B_SZ * T_SZ * NKV * 64) / TB, TB, 0, stream>>>(qkvf, cosb, sinb, kbuf);
  conv_v_kernel<<<(B_SZ * T_SZ * NKV * DH) / TB, TB, 0, stream>>>(qkvf, vtb);

  // 4. sliding-window flash attention (one wave per 16-query tile)
  attn_kernel<<<(B_SZ * NH * (T_SZ / 16)) / 8, TB, 0, stream>>>(qbuf, kbuf, vtb, ybuf);

  // 5. output projection: [4096,2048] x [2048,2048] -> d_out (f32)
  gemm_bf16_kernel<<<(NROWS / 128) * (DM / 128), TB, 0, stream>>>(
      ybuf, wot, out, NROWS, DM, DM);
}
